// BindingSiteGCN_49735721288423
// MI455X (gfx1250) — matrix operand order repaired and checked by reading, not compiled, not run on hardware
//
#include <hip/hip_runtime.h>

// ---------------------------------------------------------------------------
// CDNA5 (gfx1250) GCN pipeline: bf16 WMMA GEMMs with async-LDS-staged weights
// + f32 atomic scatter-add aggregation.
// ---------------------------------------------------------------------------

typedef __attribute__((ext_vector_type(16))) __bf16 v16bf;
typedef __attribute__((ext_vector_type(8)))  float  v8f;
typedef __attribute__((ext_vector_type(4)))  int    v4i;

#define NEG_SLOPE 0.15f

#if defined(__HIP_DEVICE_COMPILE__) && __has_builtin(__builtin_amdgcn_global_load_async_to_lds_b128)
#define HAVE_ASYNC_LDS 1
#endif

__device__ __forceinline__ unsigned short f2bf(float f) {
    unsigned int u = __float_as_uint(f);
    u += 0x7fffu + ((u >> 16) & 1u);     // round-to-nearest-even
    return (unsigned short)(u >> 16);
}

__device__ __forceinline__ float lrelu(float v) {
    return v > 0.0f ? v : v * NEG_SLOPE;
}

__device__ __forceinline__ void atomAddF(float* p, float v) {
    unsafeAtomicAdd(p, v);               // global_atomic_add_f32
}

// 16-byte global -> LDS copy; async (ASYNCcnt) when the gfx1250 builtin exists.
__device__ __forceinline__ void cp16_to_lds(unsigned short* lds, const uint4* g) {
#ifdef HAVE_ASYNC_LDS
    __builtin_amdgcn_global_load_async_to_lds_b128(
        (__attribute__((address_space(1))) v4i*)(v4i*)g,
        (__attribute__((address_space(3))) v4i*)(v4i*)lds, 0, 0);
#else
    *reinterpret_cast<uint4*>(lds) = *g;   // vmem load + ds_store fallback
#endif
}

__device__ __forceinline__ void cp_wait() {
#ifdef HAVE_ASYNC_LDS
#if __has_builtin(__builtin_amdgcn_s_wait_asynccnt)
    __builtin_amdgcn_s_wait_asynccnt(0);
#else
    asm volatile("s_wait_asynccnt 0" ::: "memory");
#endif
#endif
}

// ---------------- degree / normalization ----------------------------------

__global__ void k_deg_init(float* deg, int N) {
    int i = blockIdx.x * blockDim.x + threadIdx.x;
    if (i < N) deg[i] = 1.0f;            // self-loop contributes 1
}

__global__ void k_deg_edge(const int* __restrict__ dst, float* deg, int E) {
    int i = blockIdx.x * blockDim.x + threadIdx.x;
    if (i < E) atomAddF(&deg[dst[i]], 1.0f);
}

__global__ void k_dis(const float* __restrict__ deg, float* dis, int N) {
    int i = blockIdx.x * blockDim.x + threadIdx.x;
    if (i < N) {
        float d = deg[i];
        dis[i] = d > 0.0f ? rsqrtf(fmaxf(d, 1.0f)) : 0.0f;
    }
}

// ---------------- bf16 conversion helpers ---------------------------------

__global__ void k_cvt_bf16(const float* __restrict__ x, unsigned short* __restrict__ o,
                           long long n) {
    long long i = (long long)blockIdx.x * blockDim.x + threadIdx.x;
    if (i < n) o[i] = f2bf(x[i]);
}

// Wt[n*fi + k] = bf16(W[k*fo + n])  (transpose so B-frag loads are contiguous)
__global__ void k_wt_bf16(const float* __restrict__ W, unsigned short* __restrict__ Wt,
                          int fi, int fo) {
    int gid = blockIdx.x * blockDim.x + threadIdx.x;
    if (gid >= fi * fo) return;
    int n = gid / fi;
    int k = gid - n * fi;
    Wt[gid] = f2bf(W[(size_t)k * fo + n]);
}

// ---------------- WMMA bf16 GEMM: C[M,Nout] = A[M,K] * Bt[Nout,K]^T --------
// Block = 8 waves. Each wave owns one 16-row m-tile; all 8 waves share the
// same 64-column group, whose B tile is staged through LDS (double-buffered,
// async) in 64-K chunks. A fragments are double-buffered in registers across
// chunks. Fragment layouts per CDNA5 ISA 7.12.2. Requires K % 128 == 0 or
// K/64 even (K = 128/512/256 here -> nchunks 2/8/4, always even).

#define BROW 72              // padded LDS row (bf16 elems): 144B = 36 dwords

union AFrag { uint4 u[2]; v16bf v; };

__global__ __launch_bounds__(256) void k_gemm_bf16(
        const unsigned short* __restrict__ A,
        const unsigned short* __restrict__ Bt,
        float* __restrict__ C,
        int M, int K, int Nout) {
    __shared__ unsigned short ldsB[2][64 * BROW];

    const int ntiles  = Nout >> 6;                 // 64-column groups
    const int mtiles  = (M + 15) >> 4;
    const int bid     = blockIdx.x;
    const int mb      = bid / ntiles;
    const int nbase   = (bid - mb * ntiles) << 6;

    const int tid  = threadIdx.x;
    const int wave = tid >> 5;
    const int lane = tid & 31;
    const int half = lane >> 4;                    // K-half select
    const int l15  = lane & 15;

    const int mtile = mb * 8 + wave;               // may overrun on tail block
    int mt = mtile < mtiles ? mtile : mtiles - 1;
    int row = (mt << 4) + l15;
    if (row >= M) row = M - 1;
    const unsigned short* Arow = A + (size_t)row * K;

    // staging role: 4 threads per column, each copies 2x16B (16 K-elems)
    const int scol = tid >> 2;                     // 0..63
    const int sseg = tid & 3;                      // 0..3
    const unsigned short* gB = Bt + (size_t)(nbase + scol) * K + sseg * 16;
    unsigned short* lrow0 = &ldsB[0][scol * BROW + sseg * 16];
    unsigned short* lrow1 = &ldsB[1][scol * BROW + sseg * 16];

    const int nchunks = K >> 6;                    // 64-K chunks (even)

    auto stageB = [&](unsigned short* d, int kc) {
        const uint4* s = reinterpret_cast<const uint4*>(gB + (kc << 6));
        cp16_to_lds(d,     s);
        cp16_to_lds(d + 8, s + 1);
    };
    auto loadA = [&](AFrag* dst, int kc) {
#pragma unroll
        for (int ks = 0; ks < 2; ++ks) {
            const uint4* pa = reinterpret_cast<const uint4*>(
                Arow + (kc << 6) + (ks << 5) + half * 8);
            dst[ks].u[0] = pa[0];     // K = base .. base+7   (VGPR 0-3)
            dst[ks].u[1] = pa[2];     // K = base+16..base+23 (VGPR 4-7)
        }
    };

    v8f acc[4];
    acc[0] = v8f{}; acc[1] = v8f{}; acc[2] = v8f{}; acc[3] = v8f{};

    auto computeChunk = [&](const AFrag* af, int buf) {
#pragma unroll
        for (int ks = 0; ks < 2; ++ks) {
            // batch all 4 B-frag ds_loads, then 4 back-to-back WMMAs
            union { uint4 u[2]; v16bf v; } bfr[4];
#pragma unroll
            for (int nt = 0; nt < 4; ++nt) {
                const uint4* pb = reinterpret_cast<const uint4*>(
                    &ldsB[buf][(nt * 16 + l15) * BROW + (ks << 5) + half * 8]);
                bfr[nt].u[0] = pb[0];
                bfr[nt].u[1] = pb[2];
            }
#pragma unroll
            for (int nt = 0; nt < 4; ++nt) {
                acc[nt] = __builtin_amdgcn_wmma_f32_16x16x32_bf16(
                    false, af[ks].v, false, bfr[nt].v, (short)0, acc[nt], false, false);
            }
        }
    };

    // prime: B chunk 0 into LDS buf 0, A chunk 0 into registers
    AFrag afA[2], afB[2];
    stageB(lrow0, 0);
    loadA(afA, 0);
    cp_wait();
    __syncthreads();

    for (int kc = 0; kc < nchunks; kc += 2) {
        // ---- chunk kc (LDS buf 0, regs afA); stage kc+1 meanwhile ----
        stageB(lrow1, kc + 1);
        loadA(afB, kc + 1);
        if (kc + 2 < nchunks)
            __builtin_prefetch(Arow + ((kc + 2) << 6), 0, 3);
        computeChunk(afA, 0);
        cp_wait();
        __syncthreads();

        // ---- chunk kc+1 (LDS buf 1, regs afB); stage kc+2 meanwhile ----
        if (kc + 2 < nchunks) {
            stageB(lrow0, kc + 2);
            loadA(afA, kc + 2);
        }
        computeChunk(afB, 1);
        cp_wait();
        __syncthreads();
    }

    // C store: f32 16x16 layout: lane = N col (l15), VGPR r -> M = r + half*8
    if (mtile < mtiles) {
        int mrow = (mtile << 4) + half * 8;
#pragma unroll
        for (int r = 0; r < 8; ++r) {
            int m = mrow + r;
            if (m < M) {
                size_t base = (size_t)m * Nout + nbase + l15;
                C[base +  0] = acc[0][r];
                C[base + 16] = acc[1][r];
                C[base + 32] = acc[2][r];
                C[base + 48] = acc[3][r];
            }
        }
    }
}

// ---------------- aggregation ---------------------------------------------

// agg[i,:] = dis[i]^2 * hW[i,:]   (self-loop term; also initializes agg)
__global__ void k_selfinit(const float* __restrict__ dis,
                           const float* __restrict__ hW,
                           float* __restrict__ agg,
                           long long total4, int c4shift) {
    long long gid = (long long)blockIdx.x * blockDim.x + threadIdx.x;
    if (gid >= total4) return;
    int i  = (int)(gid >> c4shift);
    float s = dis[i];
    float n2 = s * s;
    float4 v = reinterpret_cast<const float4*>(hW)[gid];
    float4 o;
    o.x = v.x * n2; o.y = v.y * n2; o.z = v.z * n2; o.w = v.w * n2;
    reinterpret_cast<float4*>(agg)[gid] = o;
}

// agg[dst,:] += dis[src]*dis[dst] * hW[src,:]  (one thread per edge x 4 feats)
__global__ void k_scatter(const int* __restrict__ src, const int* __restrict__ dst,
                          const float* __restrict__ dis,
                          const float* __restrict__ hW, float* __restrict__ agg,
                          long long total, int Dout, int c4shift) {
    long long gid = (long long)blockIdx.x * blockDim.x + threadIdx.x;
    if (gid >= total) return;
    int e = (int)(gid >> c4shift);
    int c = (int)gid & ((Dout >> 2) - 1);
    int s = src[e], d = dst[e];
    float nrm = dis[s] * dis[d];
    float4 v = reinterpret_cast<const float4*>(hW + (size_t)s * Dout)[c];
    float* o = agg + (size_t)d * Dout + (c << 2);
    atomAddF(o + 0, v.x * nrm);
    atomAddF(o + 1, v.y * nrm);
    atomAddF(o + 2, v.z * nrm);
    atomAddF(o + 3, v.w * nrm);
}

// agg = lrelu(agg + b) in place; also emit bf16 copy for the next GEMM
__global__ void k_finalize(float* __restrict__ agg, const float* __restrict__ bias,
                           unsigned short* __restrict__ nxt, long long total, int Dout) {
    long long gid = (long long)blockIdx.x * blockDim.x + threadIdx.x;
    if (gid >= total) return;
    int c = (int)gid & (Dout - 1);
    float v = lrelu(agg[gid] + bias[c]);
    agg[gid] = v;
    nxt[gid] = f2bf(v);
}

// ---------------- head: pre_fc (128->16), MLP (16->32->2) ------------------

__global__ void k_prefc(const float* __restrict__ h3, const float* __restrict__ Wp,
                        const float* __restrict__ bp, float* __restrict__ h16, int N) {
    int gid = blockIdx.x * blockDim.x + threadIdx.x;
    if (gid >= N * 16) return;
    int node = gid >> 4;
    int o    = gid & 15;
    const float* hp = h3 + (size_t)node * 128;
    float s = bp[o];
#pragma unroll 8
    for (int k = 0; k < 128; ++k) s += hp[k] * Wp[k * 16 + o];
    h16[gid] = s;                                   // no activation on pre_fc
}

__global__ void k_head(const float* __restrict__ h16,
                       const float* __restrict__ Wf1, const float* __restrict__ bf1,
                       const float* __restrict__ Wf2, const float* __restrict__ bf2,
                       float* __restrict__ out, int N) {
    int node = blockIdx.x * blockDim.x + threadIdx.x;
    if (node >= N) return;
    float hi[16];
#pragma unroll
    for (int k = 0; k < 16; ++k) hi[k] = h16[(size_t)node * 16 + k];
    float o0 = bf2[0], o1 = bf2[1];
#pragma unroll
    for (int j = 0; j < 32; ++j) {
        float s = bf1[j];
#pragma unroll
        for (int k = 0; k < 16; ++k) s += hi[k] * Wf1[k * 32 + j];
        s = lrelu(s);
        o0 += s * Wf2[j * 2 + 0];
        o1 += s * Wf2[j * 2 + 1];
    }
    out[(size_t)node * 2 + 0] = o0;
    out[(size_t)node * 2 + 1] = o1;
}

// ---------------------------------------------------------------------------

static inline unsigned blks(long long n, int t) { return (unsigned)((n + t - 1) / t); }

extern "C" void kernel_launch(void* const* d_in, const int* in_sizes, int n_in,
                              void* d_out, int out_size, void* d_ws, size_t ws_size,
                              hipStream_t stream) {
    const float* x   = (const float*)d_in[0];
    const int*   ei  = (const int*)d_in[1];
    const float* W1  = (const float*)d_in[3];  const float* b1  = (const float*)d_in[4];
    const float* W2  = (const float*)d_in[5];  const float* b2  = (const float*)d_in[6];
    const float* W3  = (const float*)d_in[7];  const float* b3  = (const float*)d_in[8];
    const float* Wp  = (const float*)d_in[9];  const float* bp  = (const float*)d_in[10];
    const float* Wf1 = (const float*)d_in[11]; const float* bf1 = (const float*)d_in[12];
    const float* Wf2 = (const float*)d_in[13]; const float* bf2 = (const float*)d_in[14];

    const int N = in_sizes[0] / 128;
    const int E = in_sizes[1] / 2;
    const int* srcI = ei;
    const int* dstI = ei + E;

    // workspace carve-up (256B aligned)
    char* p = (char*)d_ws;
    auto carve = [&](size_t bytes) -> char* {
        char* r = p; p += (bytes + 255) & ~(size_t)255; return r;
    };
    float*          deg = (float*)carve((size_t)N * 4);
    float*          dis = (float*)carve((size_t)N * 4);
    unsigned short* inb = (unsigned short*)carve((size_t)N * 512 * 2);
    float*          hW  = (float*)carve((size_t)N * 512 * 4);
    float*          agg = (float*)carve((size_t)N * 512 * 4);
    float*          h16 = (float*)carve((size_t)N * 16 * 4);
    unsigned short* Wt1 = (unsigned short*)carve((size_t)128 * 512 * 2);
    unsigned short* Wt2 = (unsigned short*)carve((size_t)512 * 256 * 2);
    unsigned short* Wt3 = (unsigned short*)carve((size_t)256 * 128 * 2);

    const int T = 256;

    // symmetric normalization
    k_deg_init<<<blks(N, T), T, 0, stream>>>(deg, N);
    k_deg_edge<<<blks(E, T), T, 0, stream>>>(dstI, deg, E);
    k_dis     <<<blks(N, T), T, 0, stream>>>(deg, dis, N);

    // one-time bf16 conversions
    k_cvt_bf16<<<blks((long long)N * 128, T), T, 0, stream>>>(x, inb, (long long)N * 128);
    k_wt_bf16 <<<blks(128 * 512, T), T, 0, stream>>>(W1, Wt1, 128, 512);
    k_wt_bf16 <<<blks(512 * 256, T), T, 0, stream>>>(W2, Wt2, 512, 256);
    k_wt_bf16 <<<blks(256 * 128, T), T, 0, stream>>>(W3, Wt3, 256, 128);

    auto run_layer = [&](const unsigned short* Wt, const float* bias, int Din, int Dout) {
        int mtiles  = (N + 15) / 16;
        int mblocks = (mtiles + 7) / 8;
        int grid    = mblocks * (Dout / 64);
        k_gemm_bf16<<<grid, T, 0, stream>>>(inb, Wt, hW, N, Din, Dout);

        int sh = __builtin_ctz(Dout / 4);
        long long tot4 = (long long)N * (Dout / 4);
        k_selfinit<<<blks(tot4, T), T, 0, stream>>>(dis, hW, agg, tot4, sh);

        long long etot = (long long)E * (Dout / 4);
        k_scatter<<<blks(etot, T), T, 0, stream>>>(srcI, dstI, dis, hW, agg, etot, Dout, sh);

        long long ftot = (long long)N * Dout;
        k_finalize<<<blks(ftot, T), T, 0, stream>>>(agg, bias, inb, ftot, Dout);
    };

    run_layer(Wt1, b1, 128, 512);
    run_layer(Wt2, b2, 512, 256);
    run_layer(Wt3, b3, 256, 128);   // after this, agg[0 .. N*128) == h3 (f32)

    k_prefc<<<blks((long long)N * 16, T), T, 0, stream>>>(agg, Wp, bp, h16, N);
    k_head <<<blks(N, T), T, 0, stream>>>(h16, Wf1, bf1, Wf2, bf2, (float*)d_out, N);
}
